// Encoder_z0_ODE_ConvGRU_53584011985387
// MI455X (gfx1250) — matrix operand order repaired and checked by reading, not compiled, not run on hardware
//
#include <hip/hip_runtime.h>
#include <hip/hip_bf16.h>

typedef __attribute__((ext_vector_type(16))) __bf16 v16bf;
typedef __attribute__((ext_vector_type(8)))  __bf16 v8bf;
typedef __attribute__((ext_vector_type(8)))  float  v8f;

#define DEVINL __device__ __forceinline__

namespace cfg {
constexpr int Bn = 16, Tn = 16, Cx = 64, Hh = 32, Ww = 32, HD = 64;
constexpr int HW    = Hh * Ww;              // 1024
constexpr int NPIX  = Bn * HW;              // 16384 pixels
constexpr int STATE = NPIX * HD;            // 1048576 elems, unpadded f32 state
constexpr int Hp = Hh + 2, Wp = Ww + 2;     // zero-halo padded spatial dims
constexpr int PSTATE = Bn * Hp * Wp * HD;   // padded bf16 buffers
constexpr int ROWT  = Bn * Hh;              // 512 row tiles (32 px each)
constexpr int TILES = NPIX / 16;            // 1024 16-px tiles (head kernel)
}

// padded NHWC index: interior pixel (y,x) lives at (y+1, x+1)
DEVINL size_t pidx(int b, int yp, int xp, int c) {
  using namespace cfg;
  return ((size_t)(b * Hp + yp) * Wp + xp) * HD + c;
}

// ---------------- fragment builders (contiguous k = 16*half + s, same order for A and B) -------
// A (16x32 bf16): lane m=l%16 is row M; half=l/16 picks K groups
//   16 consecutive K at 16*half (matches B fragment)
DEVINL v16bf make_afrag(const __bf16* ap, int kb, int half) {
  v8bf lo = *(const v8bf*)(ap + kb + 16 * half);
  v8bf hi = *(const v8bf*)(ap + kb + 16 * half + 8);
  v16bf r;
#pragma unroll
  for (int i = 0; i < 8; ++i) { r[i] = lo[i]; r[8 + i] = hi[i]; }
  return r;
}

// B (32x16 bf16): lane n=l%16 is column N; 16 consecutive K at 16*(l/16)
DEVINL v16bf make_bfrag(const __bf16* bp) {
  v8bf b0 = *(const v8bf*)bp;
  v8bf b1 = *(const v8bf*)(bp + 8);
  v16bf r;
#pragma unroll
  for (int i = 0; i < 8; ++i) { r[i] = b0[i]; r[8 + i] = b1[i]; }
  return r;
}

template <int N>
DEVINL void zero_acc(v8f (&a)[N]) {
#pragma unroll
  for (int t = 0; t < N; ++t)
#pragma unroll
    for (int j = 0; j < 8; ++j) a[t][j] = 0.0f;
}

// One concat-segment of an implicit-GEMM 3x3 conv over a FULL 32-px row.
// Two M-tiles (x 0..15 and 16..31) share every B fragment: 2 WMMAs per
// weight load pair. act: zero-halo padded NHWC bf16 (SEGC ch);
// wt: [9][Cout][CinTot]; segment covers weight ci in [ciofs, ciofs+SEGC).
template <int NACC, int SEGC>
DEVINL void conv3x3_row(const __bf16* __restrict__ act,
                        const __bf16* __restrict__ wt,
                        int Cout, int CinTot, int ciofs,
                        int b, int y, int lane,
                        v8f (&acc0)[NACC], v8f (&acc1)[NACC]) {
  using namespace cfg;
  const int m = lane & 15, half = lane >> 4;
  const __bf16* arow = act + ((size_t)(b * Hp + y) * Wp + m) * SEGC;
#pragma unroll
  for (int ky = 0; ky < 3; ++ky) {
#pragma unroll
    for (int kx = 0; kx < 3; ++kx) {
      const __bf16* ap = arow + ((size_t)ky * Wp + kx) * SEGC;
      const __bf16* wp = wt + (size_t)(ky * 3 + kx) * Cout * CinTot + ciofs;
      __builtin_prefetch(wp + (size_t)Cout * CinTot, 0, 1);  // next tap weights
#pragma unroll
      for (int kb = 0; kb < SEGC; kb += 32) {
        v16bf af0 = make_afrag(ap, kb, half);                // pixels x=0..15
        v16bf af1 = make_afrag(ap + 16 * SEGC, kb, half);    // pixels x=16..31
#pragma unroll
        for (int t = 0; t < NACC; ++t) {
          const __bf16* bp = wp + (size_t)(t * 16 + m) * CinTot + kb + 16 * half;
          v16bf bf = make_bfrag(bp);
          acc0[t] = __builtin_amdgcn_wmma_f32_16x16x32_bf16(
              false, af0, false, bf, (short)0, acc0[t], false, false);
          acc1[t] = __builtin_amdgcn_wmma_f32_16x16x32_bf16(
              false, af1, false, bf, (short)0, acc1[t], false, false);
        }
      }
    }
  }
}

// ---------------- prep kernels ---------------------------------------------
// input (B,T,C,H,W) f32 -> step slice padded (B,H+2,W+2,C) bf16 interior
__global__ void prep_xs_kernel(const float* __restrict__ x, int s,
                               __bf16* __restrict__ xs) {
  using namespace cfg;
  int tid = blockIdx.x * blockDim.x + threadIdx.x;   // STATE threads
  int c = tid & 63;
  int p = tid >> 6;
  int b = p >> 10, pix = p & 1023;
  int y = pix >> 5, xx = pix & 31;
  size_t src = (((size_t)b * Tn + (Tn - 1 - s)) * Cx + c) * HW + pix;
  xs[pidx(b, y + 1, xx + 1, c)] = (__bf16)x[src];
}

// OIHW(taps) f32 -> [tap][o][i] bf16
__global__ void prep_w_kernel(const float* __restrict__ src,
                              __bf16* __restrict__ dst, int O, int I, int ntap,
                              int total) {
  int tid = blockIdx.x * blockDim.x + threadIdx.x;
  if (tid >= total) return;
  int i = tid % I;
  int rest = tid / I;
  int o = rest % O;
  int tap = rest / O;
  dst[tid] = (__bf16)src[((size_t)o * I + i) * ntap + tap];
}

// zero h0 (f32) and all padded bf16 activation buffers (incl. halos)
__global__ void zero_bufs_kernel(float* hf, __bf16* xs, __bf16* hb,
                                 __bf16* hodeb, __bf16* rhb) {
  using namespace cfg;
  int tid = blockIdx.x * blockDim.x + threadIdx.x;   // PSTATE threads
  const __bf16 z = (__bf16)0.0f;
  xs[tid] = z; hb[tid] = z; hodeb[tid] = z; rhb[tid] = z;
  if (tid < STATE) hf[tid] = 0.0f;
}

// ---------------- per-step kernels (one 32-px row per wave) -----------------
// h_ode = h + tanh(conv3x3(h, w_ode) + b_ode) * dt
__global__ void ode_kernel(const __bf16* __restrict__ hb,
                           const float* __restrict__ hf,
                           const __bf16* __restrict__ woT,
                           const float* __restrict__ b_ode,
                           const float* __restrict__ ts, int s,
                           float* __restrict__ hodef,
                           __bf16* __restrict__ hodeb) {
  using namespace cfg;
  const int lane = threadIdx.x & 31;
  const int tid = (blockIdx.x * blockDim.x + threadIdx.x) >> 5;
  const int b = tid >> 5, y = tid & 31;
  v8f acc0[4], acc1[4]; zero_acc(acc0); zero_acc(acc1);
  conv3x3_row<4, 64>(hb, woT, 64, 64, 0, b, y, lane, acc0, acc1);
  const float dt = (s == 0) ? -0.01f : (ts[Tn - 1 - s] - ts[Tn - s]);
  const int m = lane & 15, half = lane >> 4;
#pragma unroll
  for (int pt = 0; pt < 2; ++pt) {
#pragma unroll
    for (int t = 0; t < 4; ++t) {
      const int c = t * 16 + m;
      const float bias = b_ode[c];
#pragma unroll
      for (int j = 0; j < 8; ++j) {
        const float a = (pt == 0) ? acc0[t][j] : acc1[t][j];
        const int px = pt * 16 + j + 8 * half;
        const size_t idx = ((size_t)(b * Hh + y) * Ww + px) * HD + c;
        const float ho = hf[idx] + tanhf(a + bias) * dt;
        hodef[idx] = ho;
        hodeb[pidx(b, y + 1, px + 1, c)] = (__bf16)ho;
      }
    }
  }
}

// gates = sigmoid(conv3x3([x, h_ode], w_gates) + b); rh = reset*h_ode; keep update
__global__ void gates_kernel(const __bf16* __restrict__ xs,
                             const __bf16* __restrict__ hodeb,
                             const float* __restrict__ hodef,
                             const __bf16* __restrict__ wgT,
                             const float* __restrict__ b_gates,
                             __bf16* __restrict__ rhb,
                             float* __restrict__ updf) {
  using namespace cfg;
  const int lane = threadIdx.x & 31;
  const int tid = (blockIdx.x * blockDim.x + threadIdx.x) >> 5;
  const int b = tid >> 5, y = tid & 31;
  v8f acc0[8], acc1[8]; zero_acc(acc0); zero_acc(acc1);
  conv3x3_row<8, 64>(xs,    wgT, 128, 128, 0,  b, y, lane, acc0, acc1);
  conv3x3_row<8, 64>(hodeb, wgT, 128, 128, 64, b, y, lane, acc0, acc1);
  const int m = lane & 15, half = lane >> 4;
#pragma unroll
  for (int pt = 0; pt < 2; ++pt) {
#pragma unroll
    for (int t = 0; t < 8; ++t) {
      const int c = t * 16 + m;
      const float bias = b_gates[c];
#pragma unroll
      for (int j = 0; j < 8; ++j) {
        const float a = (pt == 0) ? acc0[t][j] : acc1[t][j];
        const int px = pt * 16 + j + 8 * half;
        const size_t sidx = ((size_t)(b * Hh + y) * Ww + px) * HD;
        const float g = 1.0f / (1.0f + __expf(-(a + bias)));
        if (t < 4) rhb[pidx(b, y + 1, px + 1, c)] = (__bf16)(g * hodef[sidx + c]);
        else       updf[sidx + (c - 64)] = g;
      }
    }
  }
}

// cand = tanh(conv3x3([x, rh], w_can)+b); h = mask*((1-u)*h_ode+u*cand)+(1-mask)*h_ode
__global__ void can_kernel(const __bf16* __restrict__ xs,
                           const __bf16* __restrict__ rhb,
                           const float* __restrict__ hodef,
                           const float* __restrict__ updf,
                           const __bf16* __restrict__ wcT,
                           const float* __restrict__ b_can,
                           const float* __restrict__ mask, int s,
                           float* __restrict__ hf, __bf16* __restrict__ hb) {
  using namespace cfg;
  const int lane = threadIdx.x & 31;
  const int tid = (blockIdx.x * blockDim.x + threadIdx.x) >> 5;
  const int b = tid >> 5, y = tid & 31;
  v8f acc0[4], acc1[4]; zero_acc(acc0); zero_acc(acc1);
  conv3x3_row<4, 64>(xs,  wcT, 64, 128, 0,  b, y, lane, acc0, acc1);
  conv3x3_row<4, 64>(rhb, wcT, 64, 128, 64, b, y, lane, acc0, acc1);
  const float mv = mask[b * Tn + (Tn - 1 - s)];
  const int m = lane & 15, half = lane >> 4;
#pragma unroll
  for (int pt = 0; pt < 2; ++pt) {
#pragma unroll
    for (int t = 0; t < 4; ++t) {
      const int c = t * 16 + m;
      const float bias = b_can[c];
#pragma unroll
      for (int j = 0; j < 8; ++j) {
        const float a = (pt == 0) ? acc0[t][j] : acc1[t][j];
        const int px = pt * 16 + j + 8 * half;
        const size_t idx = ((size_t)(b * Hh + y) * Ww + px) * HD + c;
        const float cand = tanhf(a + bias);
        const float ho = hodef[idx], u = updf[idx];
        const float hn = (1.0f - u) * ho + u * cand;
        const float hx = mv * hn + (1.0f - mv) * ho;
        hf[idx] = hx;
        hb[pidx(b, y + 1, px + 1, c)] = (__bf16)hx;
      }
    }
  }
}

// head: z = relu(h @ w1 + b1); out = z @ w2 + b2; mean=out[:64], std=|out[64:]|
__global__ void head_kernel(const __bf16* __restrict__ hb,
                            const __bf16* __restrict__ w1T,
                            const float* __restrict__ b1,
                            const __bf16* __restrict__ w2T,
                            const float* __restrict__ b2,
                            float* __restrict__ out) {
  using namespace cfg;
  __shared__ __align__(16) __bf16 zsh[8][16][64];
  const int lane = threadIdx.x & 31;
  const int w = threadIdx.x >> 5;
  const int tid = blockIdx.x * 8 + w;
  const int b = tid >> 6;
  const int r = tid & 63;
  const int y = r >> 1, x0 = (r & 1) << 4;
  const int m = lane & 15, half = lane >> 4;

  // GEMM1: 16px x 64ch, K=64
  v8f acc[4]; zero_acc(acc);
  const __bf16* aprow = hb + pidx(b, y + 1, x0 + m + 1, 0);
#pragma unroll
  for (int kb = 0; kb < 64; kb += 32) {
    v16bf af = make_afrag(aprow, kb, half);
#pragma unroll
    for (int t = 0; t < 4; ++t) {
      const __bf16* bp = w1T + (size_t)(t * 16 + m) * 64 + kb + 16 * half;
      acc[t] = __builtin_amdgcn_wmma_f32_16x16x32_bf16(
          false, af, false, make_bfrag(bp), (short)0, acc[t], false, false);
    }
  }
#pragma unroll
  for (int t = 0; t < 4; ++t) {
    const int c = t * 16 + m;
    const float bias = b1[c];
#pragma unroll
    for (int j = 0; j < 8; ++j) {
      float z = acc[t][j] + bias;
      zsh[w][j + 8 * half][c] = (__bf16)(z > 0.0f ? z : 0.0f);
    }
  }
  __syncthreads();

  // GEMM2: 16px x 128ch, K=64, A from LDS
  v8f acc2[8]; zero_acc(acc2);
  const __bf16* zp = &zsh[w][m][0];
#pragma unroll
  for (int kb = 0; kb < 64; kb += 32) {
    v16bf af = make_afrag(zp, kb, half);
#pragma unroll
    for (int t = 0; t < 8; ++t) {
      const __bf16* bp = w2T + (size_t)(t * 16 + m) * 64 + kb + 16 * half;
      acc2[t] = __builtin_amdgcn_wmma_f32_16x16x32_bf16(
          false, af, false, make_bfrag(bp), (short)0, acc2[t], false, false);
    }
  }
#pragma unroll
  for (int t = 0; t < 8; ++t) {
    const int c = t * 16 + m;
    const float bias = b2[c];
#pragma unroll
    for (int j = 0; j < 8; ++j) {
      const int px = x0 + j + 8 * half;
      const float v = acc2[t][j] + bias;
      if (t < 4)
        out[(((size_t)b * HD + c) * Hh + y) * Ww + px] = v;                        // mean
      else
        out[(size_t)STATE + (((size_t)b * HD + (c - 64)) * Hh + y) * Ww + px] = fabsf(v);  // std
    }
  }
}

// ---------------- host side -------------------------------------------------
extern "C" void kernel_launch(void* const* d_in, const int* in_sizes, int n_in,
                              void* d_out, int out_size, void* d_ws, size_t ws_size,
                              hipStream_t stream) {
  using namespace cfg;
  (void)in_sizes; (void)n_in; (void)out_size; (void)ws_size;
  const float* input   = (const float*)d_in[0];
  const float* ts      = (const float*)d_in[1];
  const float* mask    = (const float*)d_in[2];
  const float* w_gates = (const float*)d_in[3];
  const float* b_gates = (const float*)d_in[4];
  const float* w_can   = (const float*)d_in[5];
  const float* b_can   = (const float*)d_in[6];
  const float* w_ode   = (const float*)d_in[7];
  const float* b_ode   = (const float*)d_in[8];
  const float* w_t1    = (const float*)d_in[9];
  const float* b_t1    = (const float*)d_in[10];
  const float* w_t2    = (const float*)d_in[11];
  const float* b_t2    = (const float*)d_in[12];
  float* out = (float*)d_out;

  char* p = (char*)d_ws;
  auto alloc = [&](size_t bytes) {
    char* r = p;
    p += (bytes + 255) & ~(size_t)255;
    return r;
  };
  __bf16* xs    = (__bf16*)alloc((size_t)PSTATE * 2);  // padded bf16 buffers
  __bf16* hb    = (__bf16*)alloc((size_t)PSTATE * 2);
  __bf16* hodeb = (__bf16*)alloc((size_t)PSTATE * 2);
  __bf16* rhb   = (__bf16*)alloc((size_t)PSTATE * 2);
  float*  hf    = (float*) alloc((size_t)STATE * 4);   // unpadded f32 buffers
  float*  hodef = (float*) alloc((size_t)STATE * 4);
  float*  updf  = (float*) alloc((size_t)STATE * 4);
  __bf16* wgT   = (__bf16*)alloc((size_t)9 * 128 * 128 * 2);
  __bf16* wcT   = (__bf16*)alloc((size_t)9 * 64 * 128 * 2);
  __bf16* woT   = (__bf16*)alloc((size_t)9 * 64 * 64 * 2);
  __bf16* w1T   = (__bf16*)alloc((size_t)64 * 64 * 2);
  __bf16* w2T   = (__bf16*)alloc((size_t)128 * 64 * 2);

  const dim3 blk(256);
  const dim3 gridRow(ROWT / 8);                   // conv kernels: 1 row/wave
  const dim3 gridHead(TILES / 8);                 // head: 16-px tiles
  const dim3 gridElem(STATE / 256);
  const dim3 gridPad((PSTATE + 255) / 256);

  // per-call prep: weight layout transforms + zero halos + h0 = 0
  prep_w_kernel<<<(9 * 128 * 128 + 255) / 256, blk, 0, stream>>>(w_gates, wgT, 128, 128, 9, 9 * 128 * 128);
  prep_w_kernel<<<(9 * 64 * 128 + 255) / 256,  blk, 0, stream>>>(w_can,   wcT, 64, 128, 9, 9 * 64 * 128);
  prep_w_kernel<<<(9 * 64 * 64 + 255) / 256,   blk, 0, stream>>>(w_ode,   woT, 64, 64,  9, 9 * 64 * 64);
  prep_w_kernel<<<(64 * 64 + 255) / 256,       blk, 0, stream>>>(w_t1,    w1T, 64, 64,  1, 64 * 64);
  prep_w_kernel<<<(128 * 64 + 255) / 256,      blk, 0, stream>>>(w_t2,    w2T, 128, 64, 1, 128 * 64);
  zero_bufs_kernel<<<gridPad, blk, 0, stream>>>(hf, xs, hb, hodeb, rhb);

  for (int s = 0; s < Tn; ++s) {
    prep_xs_kernel<<<gridElem, blk, 0, stream>>>(input, s, xs);
    ode_kernel<<<gridRow, blk, 0, stream>>>(hb, hf, woT, b_ode, ts, s, hodef, hodeb);
    gates_kernel<<<gridRow, blk, 0, stream>>>(xs, hodeb, hodef, wgT, b_gates, rhb, updf);
    can_kernel<<<gridRow, blk, 0, stream>>>(xs, rhb, hodef, updf, wcT, b_can, mask, s, hf, hb);
  }
  head_kernel<<<gridHead, blk, 0, stream>>>(hb, w1T, b_t1, w2T, b_t2, out);
}
